// CNNLSTMJAAD2_13219909337290
// MI455X (gfx1250) — matrix operand order repaired and checked by reading, not compiled, not run on hardware
//
#include <hip/hip_runtime.h>
#include <hip/hip_bf16.h>
#include <math.h>

// ---------------------------------------------------------------------------
// CDNA5 (gfx1250) wave32 WMMA types
// ---------------------------------------------------------------------------
typedef __attribute__((ext_vector_type(16))) _Float16 v16h;
typedef __attribute__((ext_vector_type(8)))  float    v8f;

#define WAVES 8          // 8 waves of 32 per block
#define NF    256        // B*T frames

// ---------------------------------------------------------------------------
// 3x3 conv (pad=1) + bias + ReLU, implicit GEMM with v_wmma_f32_16x16x32_f16.
// Each wave owns a 16(oc) x 32(px) output tile (two accumulators, A reused).
// Tiles are staged in LDS *in WMMA operand order* ([lane][16 halves], 32B
// contiguous) so operand reads are clean ds_load_b128 pairs.
// Weights for K-aligned layers are brought into LDS with the CDNA5 async
// global->LDS path (global_load_async_to_lds_b128 + s_wait_asynccnt).
// X: [NF, Cin, H, W]  Wt: [Cout, Cin, 3, 3]  Y: [NF, Cout, H, W]
// grid: (ceil(H*W/256), Cout/16, NF), block: 256
// ---------------------------------------------------------------------------
__global__ __launch_bounds__(256) void conv3x3_wmma(
    const float* __restrict__ X, const float* __restrict__ Wt,
    const float* __restrict__ Bias, float* __restrict__ Y,
    int Cin, int Cout, int H, int W)
{
    __shared__ __align__(32) _Float16 ldsA[32 * 16];            // A operand (block-shared)
    __shared__ __align__(32) _Float16 ldsB[WAVES][2][32 * 16];  // 2 B operands per wave
    __shared__ __align__(16) float    ldsAraw[16 * 32];         // raw f32 weight staging

    const int tid    = threadIdx.x;
    const int lane   = tid & 31;
    const int wave   = tid >> 5;
    const int frame  = blockIdx.z;
    const int octile = blockIdx.y;
    const int npix   = H * W;
    const int ptile0 = (blockIdx.x * WAVES + wave) * 32;  // 32 px per wave
    const int K      = Cin * 9;
    const size_t xbase = (size_t)frame * Cin * npix;
    const bool  kAligned = (K & 31) == 0;

    // CDNA5 speculative prefetch of this block's weight slab into GL2
    __builtin_prefetch(Wt + (size_t)octile * 16 * K, 0, 0);

    v8f acc0 = {}, acc1 = {};

    for (int kk = 0; kk < K; kk += 32) {
        __syncthreads();   // previous iteration's operand reads are done

        // ---- stage raw A (f32 weights, [16 m][32 k]) -----------------------
        if (kAligned) {
            if (wave == 0) {
                #pragma unroll
                for (int it = 0; it < 4; ++it) {
                    int e   = it * 32 + lane;       // 128 segments of 16B
                    int m   = e >> 3, seg = e & 7;
                    unsigned lds_off = (unsigned)(size_t)&ldsAraw[m * 32 + seg * 4];
                    unsigned long long gaddr = (unsigned long long)
                        (Wt + (size_t)(octile * 16 + m) * K + kk + seg * 4);
                    asm volatile("global_load_async_to_lds_b128 %0, %1, off"
                                 :: "v"(lds_off), "v"(gaddr) : "memory");
                }
                asm volatile("s_wait_asynccnt 0x0" ::: "memory");
            }
        } else {
            for (int i = tid; i < 512; i += 256) {
                int m = i >> 5, k = i & 31;
                int gk = kk + k;
                ldsAraw[i] = (gk < K)
                    ? Wt[(size_t)(octile * 16 + m) * K + gk] : 0.0f;
            }
        }
        __syncthreads();   // ldsAraw visible to everyone

        // ---- permute + convert A into WMMA operand order -------------------
        // A lane = m + 16*hi holds k = {hi*8 + 0..7, 16 + hi*8 + 0..7}
        for (int i = tid; i < 512; i += 256) {
            int m  = i >> 5, k = i & 31;
            int hi = (k >> 3) & 1;
            int j  = ((k >> 4) << 3) | (k & 7);
            ldsA[(m + 16 * hi) * 16 + j] = (_Float16)ldsAraw[m * 32 + k];
        }

        // ---- im2col B tiles, written straight into operand order -----------
        // B lane = n + 16*((k>>3)&1) holds j = ((k>>4)<<3)|(k&7). Loads are unconditional from
        // clamped addresses; out-of-bounds values select to zero afterwards.
        #pragma unroll
        for (int s = 0; s < 2; ++s) {
            #pragma unroll
            for (int it = 0; it < 16; ++it) {
                int e  = it * 32 + lane;          // 0..511
                int k  = e >> 4;
                int n  = e & 15;
                int gk = kk + k;
                int p  = ptile0 + s * 16 + n;
                int gkc = (gk < K) ? gk : 0;
                int pc  = (p < npix) ? p : 0;
                int cin = gkc / 9, r = gkc % 9;
                int ky  = r / 3,   kx = r % 3;
                int y   = pc / W,  x  = pc % W;
                int iy  = y + ky - 1, ix = x + kx - 1;
                bool inb = (gk < K) && (p < npix) &&
                           (iy >= 0) && (iy < H) && (ix >= 0) && (ix < W);
                int iyc = iy < 0 ? 0 : (iy >= H ? H - 1 : iy);
                int ixc = ix < 0 ? 0 : (ix >= W ? W - 1 : ix);
                float v = X[xbase + (size_t)cin * npix + (size_t)iyc * W + ixc];
                v = inb ? v : 0.0f;
                ldsB[wave][s][(n + 16 * ((k >> 3) & 1)) * 16 + (((k >> 4) << 3) | (k & 7))] = (_Float16)v;
            }
        }
        __syncthreads();   // ldsA (block-wide) visible

        // ---- operand reads: 32B contiguous per lane -> ds_load_b128 x2 -----
        v16h a  = *(const v16h*)&ldsA[lane * 16];
        v16h bb0 = *(const v16h*)&ldsB[wave][0][lane * 16];
        v16h bb1 = *(const v16h*)&ldsB[wave][1][lane * 16];
        acc0 = __builtin_amdgcn_wmma_f32_16x16x32_f16(false, a, false, bb0,
                                                      (short)0, acc0, false, false);
        acc1 = __builtin_amdgcn_wmma_f32_16x16x32_f16(false, a, false, bb1,
                                                      (short)0, acc1, false, false);
    }

    // ---- write D (C/D layout: lane&15 = N(px), vgpr v + 8*hi = M(oc)) ------
    const int mn = lane & 15;
    const int hi = (lane >> 4) & 1;
    const size_t ybase = (size_t)frame * Cout * npix;
    #pragma unroll
    for (int s = 0; s < 2; ++s) {
        int p = ptile0 + s * 16 + mn;
        if (p < npix) {
            const v8f& acc = s ? acc1 : acc0;
            #pragma unroll
            for (int v = 0; v < 8; ++v) {
                int oc  = octile * 16 + v + hi * 8;
                float r = acc[v] + Bias[oc];
                Y[ybase + (size_t)oc * npix + p] = (r > 0.0f) ? r : 0.0f;
            }
        }
    }
}

// ---------------------------------------------------------------------------
// 2x2 / stride 2 max-pool.  X: [NF, C, H, W] -> Y: [NF, C, H/2, W/2]
// ---------------------------------------------------------------------------
__global__ void maxpool2x2(const float* __restrict__ X, float* __restrict__ Y,
                           int C, int H, int W, int total)
{
    int i = blockIdx.x * blockDim.x + threadIdx.x;
    if (i >= total) return;
    int Ho = H >> 1, Wo = W >> 1;
    int x = i % Wo; int t = i / Wo;
    int y = t % Ho; t /= Ho;
    int c = t % C;  int f = t / C;
    const float* p = X + (((size_t)f * C + c) * H + y * 2) * W + x * 2;
    float m = fmaxf(fmaxf(p[0], p[1]), fmaxf(p[W], p[W + 1]));
    Y[i] = m;
}

// ---------------------------------------------------------------------------
// gates_in = relu(feats) @ Wih^T + bih + bhh, straight WMMA GEMM.
// A: [M,K] feats (ReLU fused), Bw: Wih [N,K] row-major, D: [M,N]
// grid: (M/16, N/16), block: 32 (one wave)
// ---------------------------------------------------------------------------
__global__ __launch_bounds__(32) void gemm_relu_wmma(
    const float* __restrict__ A, const float* __restrict__ Bw,
    const float* __restrict__ b0, const float* __restrict__ b1,
    float* __restrict__ D, int M, int N, int K)
{
    const int lane  = threadIdx.x;
    const int mtile = blockIdx.x, ntile = blockIdx.y;
    const int mn    = lane & 15;
    const int hi    = (lane >> 4) & 1;
    const int row   = mtile * 16 + mn;   // A row (M index for this lane)
    const int coln  = ntile * 16 + mn;   // Bw row (N index for this lane)

    v8f acc = {};
    for (int kk = 0; kk < K; kk += 32) {
        v16h a, b;
        #pragma unroll
        for (int j = 0; j < 16; ++j) {
            int pr = j >> 1, off = j & 1;
            int ka = ((pr < 4) ? (pr * 2) : (16 + (pr - 4) * 2)) + hi * 8 + off;
            float av = A[(size_t)row * K + kk + ka];
            a[j] = (_Float16)((av > 0.0f) ? av : 0.0f);
            int kb = ((j >> 3) << 4) + hi * 8 + (j & 7);
            b[j] = (_Float16)Bw[(size_t)coln * K + kk + kb];
        }
        acc = __builtin_amdgcn_wmma_f32_16x16x32_f16(false, a, false, b,
                                                     (short)0, acc, false, false);
    }
    #pragma unroll
    for (int v = 0; v < 8; ++v) {
        int m = mtile * 16 + v + hi * 8;
        int n = ntile * 16 + mn;
        D[(size_t)m * N + n] = acc[v] + b0[n] + b1[n];
    }
}

// ---------------------------------------------------------------------------
// Sequential LSTM over T=32, B=8, H=32. gates_in rows are b*T + t, gate order
// i,f,g,o (PyTorch). One block: thread = (b, unit). hs: [T, B, 32]
// ---------------------------------------------------------------------------
__global__ __launch_bounds__(256) void lstm_kernel(
    const float* __restrict__ gates_in, const float* __restrict__ Whh,
    float* __restrict__ hs)
{
    __shared__ float hsm[8][32];
    __shared__ float csm[8][32];
    const int tid = threadIdx.x;
    const int u = tid & 31, b = tid >> 5;
    hsm[b][u] = 0.0f; csm[b][u] = 0.0f;
    __syncthreads();

    for (int t = 0; t < 32; ++t) {
        const float* g = gates_in + (size_t)(b * 32 + t) * 128;
        float gi = g[u], gf = g[32 + u], gg = g[64 + u], go = g[96 + u];
        #pragma unroll 4
        for (int k = 0; k < 32; ++k) {
            float hk = hsm[b][k];
            gi += hk * Whh[(0 * 32 + u) * 32 + k];
            gf += hk * Whh[(1 * 32 + u) * 32 + k];
            gg += hk * Whh[(2 * 32 + u) * 32 + k];
            go += hk * Whh[(3 * 32 + u) * 32 + k];
        }
        float si = 1.0f / (1.0f + __expf(-gi));
        float sf = 1.0f / (1.0f + __expf(-gf));
        float so = 1.0f / (1.0f + __expf(-go));
        float tg = tanhf(gg);
        float cn = sf * csm[b][u] + si * tg;
        float hn = so * tanhf(cn);
        __syncthreads();
        csm[b][u] = cn;
        hsm[b][u] = hn;
        hs[((size_t)t * 8 + b) * 32 + u] = hn;
        __syncthreads();
    }
}

// ---------------------------------------------------------------------------
// Head 0 on relu(h), replicated 4x (reference returns standing 4 times).
// hs: [T, B, 32]  W0: [2, 32]  out: 4 x [B, T, 2]   (512 work items)
// ---------------------------------------------------------------------------
__global__ void head_kernel(const float* __restrict__ hs,
                            const float* __restrict__ W0,
                            const float* __restrict__ b0,
                            float* __restrict__ out)
{
    int i = blockIdx.x * blockDim.x + threadIdx.x;
    if (i >= 512) return;
    int j = i & 1;
    int t = (i >> 1) & 31;
    int b = i >> 6;
    const float* hv = hs + ((size_t)t * 8 + b) * 32;
    float s = b0[j];
    #pragma unroll 4
    for (int k = 0; k < 32; ++k) {
        float hk = hv[k];
        hk = (hk > 0.0f) ? hk : 0.0f;
        s += hk * W0[j * 32 + k];
    }
    #pragma unroll
    for (int o = 0; o < 4; ++o)
        out[(size_t)o * 512 + (b * 32 + t) * 2 + j] = s;
}

// ---------------------------------------------------------------------------
// Host-side launcher.
// d_in flat order (setup_inputs, depth-first):
//   [0]  images [8,32,3,96,32]
//   [1..26]  13 x (convW [Cout,Cin,3,3], convB [Cout])
//   [27] Wih [128,1536]  [28] Whh [128,32]  [29] bih [128]  [30] bhh [128]
//   [31..38] 4 x (headW [2,32], headB [2])
// ---------------------------------------------------------------------------
extern "C" void kernel_launch(void* const* d_in, const int* in_sizes, int n_in,
                              void* d_out, int out_size, void* d_ws, size_t ws_size,
                              hipStream_t stream) {
    (void)in_sizes; (void)n_in; (void)out_size; (void)ws_size;

    static const int cfg[18] = {64, 64, 0, 128, 128, 0, 256, 256, 256, 0,
                                512, 512, 512, 0, 512, 512, 512, 0};

    const size_t BUF = 50331648; // 256*64*96*32 floats: largest activation
    float* ws    = (float*)d_ws;
    float* bufA  = ws;
    float* bufB  = ws + BUF;
    float* gates = ws + 2 * BUF;            // [256,128]
    float* hsbuf = gates + 256 * 128;       // [32,8,32]

    const float* cur = (const float*)d_in[0];
    float* nxt = bufA;
    int Cin = 3, H = 96, Wd = 32;
    int widx = 1;

    for (int ci = 0; ci < 18; ++ci) {
        int v = cfg[ci];
        if (v == 0) {
            int total = NF * Cin * (H / 2) * (Wd / 2);
            maxpool2x2<<<(total + 255) / 256, 256, 0, stream>>>(
                cur, nxt, Cin, H, Wd, total);
            H /= 2; Wd /= 2;
        } else {
            dim3 grid((H * Wd + 32 * WAVES - 1) / (32 * WAVES), v / 16, NF);
            conv3x3_wmma<<<grid, 256, 0, stream>>>(
                cur, (const float*)d_in[widx], (const float*)d_in[widx + 1],
                nxt, Cin, v, H, Wd);
            widx += 2;
            Cin = v;
        }
        cur = nxt;
        nxt = (nxt == bufA) ? bufB : bufA;
    }
    // cur now = feats [256, 1536] (frame-major, f = c*3 + y)

    gemm_relu_wmma<<<dim3(16, 8), 32, 0, stream>>>(
        cur, (const float*)d_in[27], (const float*)d_in[29],
        (const float*)d_in[30], gates, 256, 128, 1536);

    lstm_kernel<<<1, 256, 0, stream>>>(gates, (const float*)d_in[28], hsbuf);

    head_kernel<<<2, 256, 0, stream>>>(hsbuf, (const float*)d_in[31],
                                       (const float*)d_in[32], (float*)d_out);
}
